// DistGraphAgent_50233937494357
// MI455X (gfx1250) — matrix operand order repaired and checked by reading, not compiled, not run on hardware
//
#include <hip/hip_runtime.h>
#include <math.h>

#define NQ      8
#define NGRAPH  2048
#define NPG     8
#define NNODE   (NGRAPH * NPG)   // 16384 = 2^14
#define NE_     28672
#define NSTEM_  4096
#define OUTSTEM 105
#define NCONV   6

typedef __attribute__((ext_vector_type(16))) _Float16 v16h;
typedef __attribute__((ext_vector_type(8)))  float    v8f;

__device__ __forceinline__ v8f wmma_step(v16h a, v16h b, v8f c) {
  // D = A(16x32 f16) x B(32x16 f16) + C(16x16 f32)
  return __builtin_amdgcn_wmma_f32_16x16x32_f16(false, a, false, b, (short)0, c,
                                                false, false);
}
__device__ __forceinline__ float leaky_(float x) { return x > 0.f ? x : 0.01f * x; }
// native transcendentals: v_exp_f32 / v_rcp_f32 / v_cos_f32 (TRANS ops,
// co-execute with the matrix pipe)
__device__ __forceinline__ float sigmoid_(float x) {
  return __builtin_amdgcn_rcpf(1.f + __expf(-x));
}
__device__ __forceinline__ float tanh_(float x) {
  return 1.f - 2.f * __builtin_amdgcn_rcpf(__expf(2.f * x) + 1.f);
}

// A fragment (16x32 f16) built from a row-major f32 row pointer.
// lane L: row = L&15 (caller passes that row's pointer); kb = kt*32 + ((L&16)?8:0).
// element i in 0..7  -> K = kb + i ; element i in 8..15 -> K = kb + 16 + (i-8).
__device__ __forceinline__ v16h a_frag_row(const float* row, int kb) {
  v16h a;
#pragma unroll
  for (int i = 0; i < 8; ++i) {
    a[i]     = (_Float16)row[kb + i];
    a[i + 8] = (_Float16)row[kb + 16 + i];
  }
  return a;
}
// A fragment for a concatenated [rowA(64) | rowB(64)] 128-wide row.
__device__ __forceinline__ v16h a_frag_cat(const float* rA, const float* rB, int kb) {
  v16h a;
#pragma unroll
  for (int i = 0; i < 8; ++i) {
    int k0 = kb + i, k1 = kb + 16 + i;
    a[i]     = (_Float16)(k0 < 64 ? rA[k0] : rB[k0 - 64]);
    a[i + 8] = (_Float16)(k1 < 64 ? rA[k1] : rB[k1 - 64]);
  }
  return a;
}
// B fragment from fragment-packed weights: block (nt*nKt + kt) of 512 halfs,
// lane L reads 16 contiguous halfs (32B aligned).
__device__ __forceinline__ v16h b_frag(const _Float16* pw, int nKt, int kt, int nt) {
  const int lane = threadIdx.x & 31;
  return *(const v16h*)(pw + (((size_t)(nt * nKt + kt)) << 9) + (lane << 4));
}

// ---- pack f32 weight (KxN, or NxK if transposed) into B-fragment layout ----
__global__ void pack_w(const float* __restrict__ src, _Float16* __restrict__ dst,
                       int K, int N, int Npad, int transposed) {
  int nKt = K >> 5, nNt = Npad >> 4;
  int total = nKt * nNt * 512;
  for (int idx = blockIdx.x * blockDim.x + threadIdx.x; idx < total;
       idx += gridDim.x * blockDim.x) {
    int e   = idx & 15;
    int L   = (idx >> 4) & 31;
    int blk = idx >> 9;
    int kt  = blk % nKt, nt = blk / nKt;
    int n   = nt * 16 + (L & 15);
    int k   = kt * 32 + ((L & 16) ? 8 : 0) + (e & 7) + ((e & 8) << 1);
    float v = 0.f;
    if (n < N) v = transposed ? src[n * K + k] : src[k * N + n];
    dst[idx] = (_Float16)v;
  }
}

// ---- phi: vec[q*NG+g,:] = relu(leaky(cos @ W1 + b1) @ W2 + b2) ----
__global__ void __launch_bounds__(32) phi_kernel(
    const float* __restrict__ quant, const _Float16* __restrict__ W1,
    const float* __restrict__ b1, const _Float16* __restrict__ W2,
    const float* __restrict__ b2, float* __restrict__ vec) {
  __shared__ float lds[16 * 64];
  const int lane = threadIdx.x & 31;
  const int m = lane & 15, hi = (lane >> 4) & 1;
  const int r0 = blockIdx.x * 16;
  const int r = r0 + m;
  const int q = r / NGRAPH, g = r % NGRAPH;
  const float qv = quant[g * NQ + q];
  const float PI = 3.14159265358979323846f;
  v8f acc[4] = {{}, {}, {}, {}};
#pragma unroll
  for (int kt = 0; kt < 2; ++kt) {
    const int kb = kt * 32 + hi * 8;
    v16h a;
#pragma unroll
    for (int i = 0; i < 8; ++i) {
      // |arg| <= 64*pi: native v_cos_f32 path (fast reduction) is plenty
      a[i]     = (_Float16)__cosf(PI * (float)(kb + i + 1) * qv);
      a[i + 8] = (_Float16)__cosf(PI * (float)(kb + 16 + i + 1) * qv);
    }
#pragma unroll
    for (int nt = 0; nt < 4; ++nt) acc[nt] = wmma_step(a, b_frag(W1, 2, kt, nt), acc[nt]);
  }
#pragma unroll
  for (int nt = 0; nt < 4; ++nt) {
    const int col = nt * 16 + m;
    const float bb = b1[col];
#pragma unroll
    for (int j = 0; j < 8; ++j) lds[(j + hi * 8) * 64 + col] = leaky_(acc[nt][j] + bb);
  }
  __syncthreads();
  v8f acc2[4] = {{}, {}, {}, {}};
#pragma unroll
  for (int kt = 0; kt < 2; ++kt) {
    v16h a = a_frag_row(&lds[m * 64], kt * 32 + hi * 8);
#pragma unroll
    for (int nt = 0; nt < 4; ++nt) acc2[nt] = wmma_step(a, b_frag(W2, 2, kt, nt), acc2[nt]);
  }
#pragma unroll
  for (int nt = 0; nt < 4; ++nt) {
    const int col = nt * 16 + m;
    const float bb = b2[col];
#pragma unroll
    for (int j = 0; j < 8; ++j) {
      float v = acc2[nt][j] + bb;
      vec[(size_t)(r0 + j + hi * 8) * 64 + col] = v > 0.f ? v : 0.f;
    }
  }
}

// ---- b2e: H[q,i,:] = leaky([emb_block[x_i] | vec[q,batch_i]] @ W1 + b1) @ W2 + b2 ----
__global__ void __launch_bounds__(32) b2e_kernel(
    const int* __restrict__ x, const int* __restrict__ batch,
    const float* __restrict__ emb_block, const float* __restrict__ vec,
    const _Float16* __restrict__ W1, const float* __restrict__ b1,
    const _Float16* __restrict__ W2, const float* __restrict__ b2,
    float* __restrict__ H) {
  __shared__ float lds[16 * 64];
  const int lane = threadIdx.x & 31;
  const int m = lane & 15, hi = (lane >> 4) & 1;
  const int r0 = blockIdx.x * 16;
  const int q = r0 >> 14;               // NNODE = 2^14
  const int i = (r0 & (NNODE - 1)) + m;
  const float* rA = emb_block + (size_t)x[i] * 64;
  const float* rB = vec + ((size_t)q * NGRAPH + batch[i]) * 64;
  v8f acc[4] = {{}, {}, {}, {}};
#pragma unroll
  for (int kt = 0; kt < 4; ++kt) {
    v16h a = a_frag_cat(rA, rB, kt * 32 + hi * 8);
#pragma unroll
    for (int nt = 0; nt < 4; ++nt) acc[nt] = wmma_step(a, b_frag(W1, 4, kt, nt), acc[nt]);
  }
#pragma unroll
  for (int nt = 0; nt < 4; ++nt) {
    const int col = nt * 16 + m;
    const float bb = b1[col];
#pragma unroll
    for (int j = 0; j < 8; ++j) lds[(j + hi * 8) * 64 + col] = leaky_(acc[nt][j] + bb);
  }
  __syncthreads();
  v8f acc2[4] = {{}, {}, {}, {}};
#pragma unroll
  for (int kt = 0; kt < 2; ++kt) {
    v16h a = a_frag_row(&lds[m * 64], kt * 32 + hi * 8);
#pragma unroll
    for (int nt = 0; nt < 4; ++nt) acc2[nt] = wmma_step(a, b_frag(W2, 2, kt, nt), acc2[nt]);
  }
#pragma unroll
  for (int nt = 0; nt < 4; ++nt) {
    const int col = nt * 16 + m;
    const float bb = b2[col];
#pragma unroll
    for (int j = 0; j < 8; ++j)
      H[(size_t)(r0 + j + hi * 8) * 64 + col] = acc2[nt][j] + bb;
  }
}

// ---- deg ----
__global__ void deg_kernel(const int* __restrict__ dst, float* __restrict__ deg) {
  int e = blockIdx.x * blockDim.x + threadIdx.x;
  if (e < NE_) atomicAdd(&deg[dst[e]], 1.f);
}

// ---- edge messages: rank-1 factorized W_e; scatter-add into agg ----
__global__ void edge_kernel(const int* __restrict__ eidx, const int* __restrict__ eai,
                            const float* __restrict__ emb_bond,
                            const float* __restrict__ H, float* __restrict__ agg) {
  int idx = blockIdx.x * blockDim.x + threadIdx.x;
  if (idx >= NE_ * NQ) return;
  int e = idx % NE_, q = idx / NE_;
  int s = eidx[e], d = eidx[NE_ + e];
  const float* ar = emb_bond + (size_t)eai[2 * e] * 64;
  const float* br = emb_bond + (size_t)eai[2 * e + 1] * 64;
  const float* orow = H + ((size_t)q * NNODE + s) * 64;
  float acc = 0.f;
#pragma unroll 8
  for (int k = 0; k < 64; ++k) acc += orow[k] * ar[k];
  float* arow = agg + ((size_t)q * NNODE + d) * 64;
#pragma unroll 8
  for (int o = 0; o < 64; ++o) atomicAdd(arow + o, acc * br[o]);
}

// ---- fused conv_root + GRU step ----
__global__ void __launch_bounds__(32) gru_kernel(
    const float* __restrict__ Hin, float* __restrict__ Hout,
    const float* __restrict__ agg, const float* __restrict__ deg,
    const _Float16* __restrict__ Wroot, const float* __restrict__ bconv,
    const _Float16* __restrict__ Wih, const _Float16* __restrict__ Whh,
    const float* __restrict__ bih, const float* __restrict__ bhh) {
  __shared__ float lds[16 * 64];
  const int lane = threadIdx.x & 31;
  const int m = lane & 15, hi = (lane >> 4) & 1;
  const int r0 = blockIdx.x * 16;
  const int i0 = r0 & (NNODE - 1);
  const float* hrow = Hin + (size_t)(r0 + m) * 64;
  v16h ah0 = a_frag_row(hrow, hi * 8);
  v16h ah1 = a_frag_row(hrow, 32 + hi * 8);
  // m = leaky(agg/deg + h @ conv_root + conv_bias)
  v8f accr[4] = {{}, {}, {}, {}};
#pragma unroll
  for (int nt = 0; nt < 4; ++nt) {
    accr[nt] = wmma_step(ah0, b_frag(Wroot, 2, 0, nt), accr[nt]);
    accr[nt] = wmma_step(ah1, b_frag(Wroot, 2, 1, nt), accr[nt]);
  }
#pragma unroll
  for (int nt = 0; nt < 4; ++nt) {
    const int col = nt * 16 + m;
    const float bb = bconv[col];
#pragma unroll
    for (int j = 0; j < 8; ++j) {
      const int rl = j + hi * 8;
      float dg = deg[i0 + rl];
      dg = dg < 1.f ? 1.f : dg;
      float av = agg[(size_t)(r0 + rl) * 64 + col] * __builtin_amdgcn_rcpf(dg);
      lds[rl * 64 + col] = leaky_(accr[nt][j] + av + bb);
    }
  }
  __syncthreads();
  v16h am0 = a_frag_row(&lds[m * 64], hi * 8);
  v16h am1 = a_frag_row(&lds[m * 64], 32 + hi * 8);
#pragma unroll
  for (int ng = 0; ng < 4; ++ng) {
    v8f gir = {}, giz = {}, gin = {}, ghr = {}, ghz = {}, ghn = {};
    gir = wmma_step(am0, b_frag(Wih, 2, 0, ng), gir);
    gir = wmma_step(am1, b_frag(Wih, 2, 1, ng), gir);
    giz = wmma_step(am0, b_frag(Wih, 2, 0, ng + 4), giz);
    giz = wmma_step(am1, b_frag(Wih, 2, 1, ng + 4), giz);
    gin = wmma_step(am0, b_frag(Wih, 2, 0, ng + 8), gin);
    gin = wmma_step(am1, b_frag(Wih, 2, 1, ng + 8), gin);
    ghr = wmma_step(ah0, b_frag(Whh, 2, 0, ng), ghr);
    ghr = wmma_step(ah1, b_frag(Whh, 2, 1, ng), ghr);
    ghz = wmma_step(ah0, b_frag(Whh, 2, 0, ng + 4), ghz);
    ghz = wmma_step(ah1, b_frag(Whh, 2, 1, ng + 4), ghz);
    ghn = wmma_step(ah0, b_frag(Whh, 2, 0, ng + 8), ghn);
    ghn = wmma_step(ah1, b_frag(Whh, 2, 1, ng + 8), ghn);
    const int col = ng * 16 + m;
    const float bir = bih[col], biz = bih[64 + col], bin = bih[128 + col];
    const float bhr = bhh[col], bhz = bhh[64 + col], bhn = bhh[128 + col];
#pragma unroll
    for (int j = 0; j < 8; ++j) {
      const int rl = j + hi * 8;
      const size_t idx = (size_t)(r0 + rl) * 64 + col;
      float hold = Hin[idx];
      float rg = sigmoid_(gir[j] + bir + ghr[j] + bhr);
      float zg = sigmoid_(giz[j] + biz + ghz[j] + bhz);
      float nv = tanh_(gin[j] + bin + rg * (ghn[j] + bhn));
      Hout[idx] = (1.f - zg) * nv + zg * hold;
    }
  }
}

// ---- stem head: rows (s,q); MLP 128->64->64->105; mean over q via atomics ----
__global__ void __launch_bounds__(32) stem_kernel(
    const int* __restrict__ stemtypes, const int* __restrict__ stems,
    const int* __restrict__ stems_batch, const int* __restrict__ node_offsets,
    const float* __restrict__ emb_stem, const float* __restrict__ H,
    const _Float16* __restrict__ W1, const float* __restrict__ b1,
    const _Float16* __restrict__ W2, const float* __restrict__ b2,
    const _Float16* __restrict__ W3, const float* __restrict__ b3,
    float* __restrict__ outStem) {
  __shared__ float lds0[16 * 64];
  __shared__ float lds1[16 * 64];
  const int lane = threadIdx.x & 31;
  const int m = lane & 15, hi = (lane >> 4) & 1;
  const int r0 = blockIdx.x * 16;
  const int r = r0 + m;
  const int s = r >> 3, q = r & 7;
  const int sidx = node_offsets[stems_batch[s]] + stems[2 * s];
  const float* rA = H + ((size_t)q * NNODE + sidx) * 64;
  const float* rB = emb_stem + (size_t)stemtypes[s] * 64;
  v8f acc[4] = {{}, {}, {}, {}};
#pragma unroll
  for (int kt = 0; kt < 4; ++kt) {
    v16h a = a_frag_cat(rA, rB, kt * 32 + hi * 8);
#pragma unroll
    for (int nt = 0; nt < 4; ++nt) acc[nt] = wmma_step(a, b_frag(W1, 4, kt, nt), acc[nt]);
  }
#pragma unroll
  for (int nt = 0; nt < 4; ++nt) {
    const int col = nt * 16 + m;
    const float bb = b1[col];
#pragma unroll
    for (int j = 0; j < 8; ++j) lds0[(j + hi * 8) * 64 + col] = leaky_(acc[nt][j] + bb);
  }
  __syncthreads();
  v8f acc2[4] = {{}, {}, {}, {}};
#pragma unroll
  for (int kt = 0; kt < 2; ++kt) {
    v16h a = a_frag_row(&lds0[m * 64], kt * 32 + hi * 8);
#pragma unroll
    for (int nt = 0; nt < 4; ++nt) acc2[nt] = wmma_step(a, b_frag(W2, 2, kt, nt), acc2[nt]);
  }
#pragma unroll
  for (int nt = 0; nt < 4; ++nt) {
    const int col = nt * 16 + m;
    const float bb = b2[col];
#pragma unroll
    for (int j = 0; j < 8; ++j) lds1[(j + hi * 8) * 64 + col] = leaky_(acc2[nt][j] + bb);
  }
  __syncthreads();
  v16h al0 = a_frag_row(&lds1[m * 64], hi * 8);
  v16h al1 = a_frag_row(&lds1[m * 64], 32 + hi * 8);
#pragma unroll
  for (int nt = 0; nt < 7; ++nt) {
    v8f a3 = {};
    a3 = wmma_step(al0, b_frag(W3, 2, 0, nt), a3);
    a3 = wmma_step(al1, b_frag(W3, 2, 1, nt), a3);
    const int col = nt * 16 + m;
    if (col < OUTSTEM) {
      const float bb = b3[col];
#pragma unroll
      for (int j = 0; j < 8; ++j) {
        const int rr = r0 + j + hi * 8;
        atomicAdd(&outStem[(size_t)(rr >> 3) * OUTSTEM + col], (a3[j] + bb) * 0.125f);
      }
    }
  }
}

// ---- mol mean-pool: mol[(g*NQ+q)*64+c] = mean over graph's 8 nodes ----
__global__ void mol_avg(const float* __restrict__ H, float* __restrict__ mol) {
  int idx = blockIdx.x * blockDim.x + threadIdx.x;
  if (idx >= NGRAPH * NQ * 64) return;
  int c = idx & 63, rr = idx >> 6;
  int q = rr & 7, g = rr >> 3;
  float sum = 0.f;
#pragma unroll
  for (int k = 0; k < NPG; ++k)
    sum += H[((size_t)q * NNODE + g * NPG + k) * 64 + c];
  mol[(size_t)rr * 64 + c] = sum * (1.f / NPG);
}

// ---- mol head: leaky(mol @ W1 + b1) @ W2(64x1) + b2, mean over q ----
__global__ void __launch_bounds__(32) mol_gemm(
    const float* __restrict__ mol, const _Float16* __restrict__ W1,
    const float* __restrict__ b1, const float* __restrict__ W2,
    const float* __restrict__ b2, float* __restrict__ outMol) {
  __shared__ float lds[16 * 64];
  const int lane = threadIdx.x & 31;
  const int m = lane & 15, hi = (lane >> 4) & 1;
  const int r0 = blockIdx.x * 16;
  const float* row = mol + (size_t)(r0 + m) * 64;
  v8f acc[4] = {{}, {}, {}, {}};
#pragma unroll
  for (int kt = 0; kt < 2; ++kt) {
    v16h a = a_frag_row(row, kt * 32 + hi * 8);
#pragma unroll
    for (int nt = 0; nt < 4; ++nt) acc[nt] = wmma_step(a, b_frag(W1, 2, kt, nt), acc[nt]);
  }
#pragma unroll
  for (int nt = 0; nt < 4; ++nt) {
    const int col = nt * 16 + m;
    const float bb = b1[col];
#pragma unroll
    for (int j = 0; j < 8; ++j) lds[(j + hi * 8) * 64 + col] = leaky_(acc[nt][j] + bb);
  }
  __syncthreads();
  if (lane < 16) {
    float v = b2[0];
#pragma unroll 8
    for (int k = 0; k < 64; ++k) v += lds[lane * 64 + k] * W2[k];
    const int g = (r0 + lane) >> 3;
    atomicAdd(&outMol[g], v * 0.125f);
  }
}

extern "C" void kernel_launch(void* const* d_in, const int* in_sizes, int n_in,
                              void* d_out, int out_size, void* d_ws, size_t ws_size,
                              hipStream_t stream) {
  (void)in_sizes; (void)n_in; (void)ws_size;
  const int*   x            = (const int*)d_in[0];
  const int*   stemtypes    = (const int*)d_in[1];
  const int*   edge_attr    = (const int*)d_in[2];
  const int*   edge_index   = (const int*)d_in[3];
  const int*   batch        = (const int*)d_in[4];
  const int*   stems        = (const int*)d_in[5];
  const int*   stems_batch  = (const int*)d_in[6];
  const int*   node_offsets = (const int*)d_in[7];
  const float* quantiles    = (const float*)d_in[8];
  const float* emb_block    = (const float*)d_in[9];
  const float* emb_stem     = (const float*)d_in[10];
  const float* emb_bond     = (const float*)d_in[11];
  const float* phi_W1 = (const float*)d_in[12];
  const float* phi_b1 = (const float*)d_in[13];
  const float* phi_W2 = (const float*)d_in[14];
  const float* phi_b2 = (const float*)d_in[15];
  const float* b2e_W1 = (const float*)d_in[16];
  const float* b2e_b1 = (const float*)d_in[17];
  const float* b2e_W2 = (const float*)d_in[18];
  const float* b2e_b2 = (const float*)d_in[19];
  const float* conv_root = (const float*)d_in[20];
  const float* conv_bias = (const float*)d_in[21];
  const float* gru_Wih = (const float*)d_in[22];
  const float* gru_Whh = (const float*)d_in[23];
  const float* gru_bih = (const float*)d_in[24];
  const float* gru_bhh = (const float*)d_in[25];
  const float* s2p_W1 = (const float*)d_in[26];
  const float* s2p_b1 = (const float*)d_in[27];
  const float* s2p_W2 = (const float*)d_in[28];
  const float* s2p_b2 = (const float*)d_in[29];
  const float* s2p_W3 = (const float*)d_in[30];
  const float* s2p_b3 = (const float*)d_in[31];
  const float* g2p_W1 = (const float*)d_in[32];
  const float* g2p_b1 = (const float*)d_in[33];
  const float* g2p_W2 = (const float*)d_in[34];
  const float* g2p_b2 = (const float*)d_in[35];

  char* base = (char*)d_ws;
  size_t off = 0;
  auto alloc = [&](size_t bytes) {
    void* p = base + off;
    off = (off + bytes + 255) & ~(size_t)255;
    return p;
  };
  auto packhalfs = [](int K, int Npad) { return (size_t)(K >> 5) * (Npad >> 4) * 512; };

  _Float16* pPhi1 = (_Float16*)alloc(packhalfs(64, 64) * 2);
  _Float16* pPhi2 = (_Float16*)alloc(packhalfs(64, 64) * 2);
  _Float16* pB1   = (_Float16*)alloc(packhalfs(128, 64) * 2);
  _Float16* pB2   = (_Float16*)alloc(packhalfs(64, 64) * 2);
  _Float16* pRoot = (_Float16*)alloc(packhalfs(64, 64) * 2);
  _Float16* pWih  = (_Float16*)alloc(packhalfs(64, 192) * 2);
  _Float16* pWhh  = (_Float16*)alloc(packhalfs(64, 192) * 2);
  _Float16* pS1   = (_Float16*)alloc(packhalfs(128, 64) * 2);
  _Float16* pS2   = (_Float16*)alloc(packhalfs(64, 64) * 2);
  _Float16* pS3   = (_Float16*)alloc(packhalfs(64, 112) * 2);
  _Float16* pG1   = (_Float16*)alloc(packhalfs(64, 64) * 2);
  float* vec = (float*)alloc((size_t)NQ * NGRAPH * 64 * 4);
  float* Ha  = (float*)alloc((size_t)NQ * NNODE * 64 * 4);
  float* Hb  = (float*)alloc((size_t)NQ * NNODE * 64 * 4);
  float* agg = (float*)alloc((size_t)NQ * NNODE * 64 * 4);
  float* mol = (float*)alloc((size_t)NGRAPH * NQ * 64 * 4);
  float* deg = (float*)alloc((size_t)NNODE * 4);

  auto pgrid = [&](int K, int Npad) { return (int)((packhalfs(K, Npad) + 255) / 256); };
  pack_w<<<pgrid(64, 64), 256, 0, stream>>>(phi_W1, pPhi1, 64, 64, 64, 0);
  pack_w<<<pgrid(64, 64), 256, 0, stream>>>(phi_W2, pPhi2, 64, 64, 64, 0);
  pack_w<<<pgrid(128, 64), 256, 0, stream>>>(b2e_W1, pB1, 128, 64, 64, 0);
  pack_w<<<pgrid(64, 64), 256, 0, stream>>>(b2e_W2, pB2, 64, 64, 64, 0);
  pack_w<<<pgrid(64, 64), 256, 0, stream>>>(conv_root, pRoot, 64, 64, 64, 0);
  pack_w<<<pgrid(64, 192), 256, 0, stream>>>(gru_Wih, pWih, 64, 192, 192, 1);
  pack_w<<<pgrid(64, 192), 256, 0, stream>>>(gru_Whh, pWhh, 64, 192, 192, 1);
  pack_w<<<pgrid(128, 64), 256, 0, stream>>>(s2p_W1, pS1, 128, 64, 64, 0);
  pack_w<<<pgrid(64, 64), 256, 0, stream>>>(s2p_W2, pS2, 64, 64, 64, 0);
  pack_w<<<pgrid(64, 112), 256, 0, stream>>>(s2p_W3, pS3, 64, OUTSTEM, 112, 0);
  pack_w<<<pgrid(64, 64), 256, 0, stream>>>(g2p_W1, pG1, 64, 64, 64, 0);

  hipMemsetAsync(deg, 0, (size_t)NNODE * 4, stream);
  hipMemsetAsync(d_out, 0, (size_t)out_size * sizeof(float), stream);
  deg_kernel<<<(NE_ + 255) / 256, 256, 0, stream>>>(edge_index + NE_, deg);

  phi_kernel<<<(NQ * NGRAPH) / 16, 32, 0, stream>>>(quantiles, pPhi1, phi_b1, pPhi2,
                                                    phi_b2, vec);
  b2e_kernel<<<(NQ * NNODE) / 16, 32, 0, stream>>>(x, batch, emb_block, vec, pB1,
                                                   b2e_b1, pB2, b2e_b2, Ha);
  float* Hin = Ha;
  float* Hout = Hb;
  for (int t = 0; t < NCONV; ++t) {
    hipMemsetAsync(agg, 0, (size_t)NQ * NNODE * 64 * 4, stream);
    edge_kernel<<<(NE_ * NQ + 255) / 256, 256, 0, stream>>>(edge_index, edge_attr,
                                                            emb_bond, Hin, agg);
    gru_kernel<<<(NQ * NNODE) / 16, 32, 0, stream>>>(Hin, Hout, agg, deg, pRoot,
                                                     conv_bias, pWih, pWhh, gru_bih,
                                                     gru_bhh);
    float* tmp = Hin; Hin = Hout; Hout = tmp;
  }

  stem_kernel<<<(NSTEM_ * NQ) / 16, 32, 0, stream>>>(
      stemtypes, stems, stems_batch, node_offsets, emb_stem, Hin, pS1, s2p_b1, pS2,
      s2p_b2, pS3, s2p_b3, (float*)d_out);
  mol_avg<<<(NGRAPH * NQ * 64 + 255) / 256, 256, 0, stream>>>(Hin, mol);
  mol_gemm<<<(NGRAPH * NQ) / 16, 32, 0, stream>>>(
      mol, pG1, g2p_b1, g2p_W2, g2p_b2, (float*)d_out + (size_t)NSTEM_ * OUTSTEM);
}
